// LegislativeGraphModel_13065290515085
// MI455X (gfx1250) — matrix operand order repaired and checked by reading, not compiled, not run on hardware
//
#include <hip/hip_runtime.h>
#include <math.h>

// ---------------------------------------------------------------------------
// Types / constants
// ---------------------------------------------------------------------------
typedef __bf16 bf16_t;
typedef __attribute__((ext_vector_type(16))) bf16_t v16bf;
typedef __attribute__((ext_vector_type(8)))  float   v8f;

#define HIDD 192
#define NHEAD 4
#define DKH  48
#define NKT  32
#define NTOT 31500

// node-type row offsets inside the packed H buffer
#define OFF_BILL 0
#define OFF_BV   2000
#define OFF_LEG  8000
#define OFF_LT   10000
#define OFF_COM  18000
#define OFF_DON  18500
#define OFF_LOB  28500

// ---------------------------------------------------------------------------
// Workspace layout (float offsets). Total ~61.7M floats ~ 247 MB.
// ---------------------------------------------------------------------------
static const long WS_H    = 0;
static const long WS_MSG  = WS_H    + (long)NTOT*HIDD;
static const long WS_QH   = WS_MSG  + (long)NTOT*HIDD;
static const long WS_KH   = WS_QH   + (long)NTOT*HIDD;
static const long WS_VH   = WS_KH   + (long)NTOT*HIDD;
static const long WS_YB   = WS_VH   + (long)NTOT*HIDD;
static const long WS_TB   = WS_YB   + (long)NTOT*HIDD;      // 8192 x 768 FFN chunk
static const long WS_XPAD = WS_TB   + 8192L*768;            // 6000 x 800 max padded input
static const long WS_ELOG = WS_XPAD + 6000L*800;            // 300000 x 4 edge logits
static const long WS_SEGM = WS_ELOG + 300000L*4;            // seg max keys (<=8000*4)
static const long WS_SEGS = WS_SEGM + 32000;                // seg sums
static const long WS_VMSG = WS_SEGS + 32000;                // 6000 x 192
static const long WS_ACC  = WS_VMSG + 6000L*HIDD;           // 2000 x 192 seg-mean acc
static const long WS_CNT  = WS_ACC  + 2000L*HIDD;           // 2048 counts
static const long WS_AQ   = WS_CNT  + 2048;                 // 10000 x 192
static const long WS_BK   = WS_AQ   + 10000L*HIDD;          // 6000 x 192
static const long WS_SIM  = WS_BK   + 6000L*HIDD;           // 1024 x 6000 chunk
static const long WS_PAB  = WS_SIM  + 1024L*6000;           // 1024 x 2000 chunk
static const long WS_SIGS = WS_PAB  + 1024L*2000;           // 2000 sigmoids
static const long WS_AV2B = WS_SIGS + 2048;                 // 6000 clipped weights
static const long WS_TID  = WS_AV2B + 6144;                 // 2000 topic ids (int)
static const long WS_WPAD = WS_TID  + 2048;                 // 800 x 192 padded proj weight

// ---------------------------------------------------------------------------
// Device helpers
// ---------------------------------------------------------------------------
__device__ inline bf16_t f2bf(float f) {
  unsigned u = __float_as_uint(f);
  unsigned r = (u + 0x7FFFu + ((u >> 16) & 1u)) >> 16;
  unsigned short s = (unsigned short)r;
  bf16_t b;
  __builtin_memcpy(&b, &s, 2);
  return b;
}
__device__ inline float gelu_exact(float x) {
  return 0.5f * x * (1.0f + erff(x * 0.70710678118654752f));
}
__device__ inline unsigned ford(float f) {
  unsigned u = __float_as_uint(f);
  return (u & 0x80000000u) ? ~u : (u | 0x80000000u);
}
__device__ inline float funord(unsigned u) {
  return __uint_as_float((u & 0x80000000u) ? (u & 0x7FFFFFFFu) : ~u);
}

#define RED_SUM(val, red, out) \
  red[threadIdx.x] = (val); __syncthreads(); \
  for (int s_ = 128; s_ > 0; s_ >>= 1) { if ((int)threadIdx.x < s_) red[threadIdx.x] += red[threadIdx.x + s_]; __syncthreads(); } \
  out = red[0]; __syncthreads();

#define RED_MAX(val, red, out) \
  red[threadIdx.x] = (val); __syncthreads(); \
  for (int s_ = 128; s_ > 0; s_ >>= 1) { if ((int)threadIdx.x < s_) red[threadIdx.x] = fmaxf(red[threadIdx.x], red[threadIdx.x + s_]); __syncthreads(); } \
  out = red[0]; __syncthreads();

// ---------------------------------------------------------------------------
// Fill / pad kernels
// ---------------------------------------------------------------------------
__global__ void fill_f32(float* p, float v, long n) {
  long i = (long)blockIdx.x * 256 + threadIdx.x;
  if (i < n) p[i] = v;
}
__global__ void fill_u32(unsigned* p, unsigned v, long n) {
  long i = (long)blockIdx.x * 256 + threadIdx.x;
  if (i < n) p[i] = v;
}
// zero-pad a KxHIDD weight to kpad x HIDD
__global__ void pad_weight(const float* __restrict__ W, float* __restrict__ WP, int K, int kpad) {
  long i = (long)blockIdx.x * 256 + threadIdx.x;
  if (i >= (long)kpad * HIDD) return;
  int k = (int)(i / HIDD);
  WP[i] = (k < K) ? W[i] : 0.f;
}

// ---------------------------------------------------------------------------
// bf16-WMMA GEMM:  C = epi( scale * (A @ B(+C if accum)) )
// A: MxK (lda), B: KxN (LDB) or NxK if TB, C: MxN (ldc)
// REQUIRES: K % 32 == 0, N % 16 == 0, lda multiple of 4 floats.
// LDB is a compile-time constant -> B fragment loads are one base pointer
// plus immediate offsets (no per-row pointer arithmetic, no spills).
// M may be ragged: A rows are clamped (out-of-range rows never stored).
// One 16(M)x64(N) tile per wave, 4 accumulators share one A fragment.
// ---------------------------------------------------------------------------
struct GemmEpi {
  int act;                    // 0 none, 1 gelu, 2 relu
  int accum;                  // preload C into accumulator
  const float* bias;          // nullable, length N
  float scale;
  const unsigned char* mask;  // nullable: sim epilogue (where(mask,v,-1e9)+wvb)
  const float* wvb;
  int mld;                    // leading dim for mask/wvb
  int rowoff;                 // global row offset for mask/wvb
};

template<int TB, int LDB>
__global__ void gemm_wmma_t(const float* __restrict__ A, int lda,
                            const float* __restrict__ B,
                            float* __restrict__ C, int ldc,
                            int M, int N, int K, GemmEpi e)
{
  int wave = threadIdx.x >> 5;
  int lane = threadIdx.x & 31;
  int tilesN = (N + 63) >> 6;            // 64-wide wave tiles
  int tilesM = (M + 15) >> 4;
  int tile = blockIdx.x * 8 + wave;
  if (tile >= tilesM * tilesN) return;   // wave-uniform
  int tm = tile / tilesN, tn = tile - tm * tilesN;
  int m0 = tm << 4, n0 = tn << 6;
  int lm = lane & 15, lhi = lane >> 4;

  int arow = m0 + lm;
  int arowc = arow < M ? arow : (M - 1);
  const float* ap = A + (long)arowc * lda + (lhi ? 16 : 0);
  const float* bp0 = TB ? (B + (long)(n0 + lm) * LDB + (lhi ? 16 : 0))
                        : (B + (long)(lhi ? 16 : 0) * LDB + n0 + lm);

  v8f acc[4];
  if (e.accum) {
#pragma unroll
    for (int j = 0; j < 4; ++j) {
      int nb = n0 + j * 16;
      if (nb < N) {
        int n = nb + lm;
#pragma unroll
        for (int g = 0; g < 8; ++g) {
          int m = m0 + g + (lhi ? 8 : 0);
          int mc = m < M ? m : (M - 1);
          acc[j][g] = C[(long)mc * ldc + n];
        }
      }
    }
  } else {
#pragma unroll
    for (int j = 0; j < 4; ++j)
#pragma unroll
      for (int g = 0; g < 8; ++g) acc[j][g] = 0.f;
  }

  for (int k0 = 0; k0 < K; k0 += 32) {
    __builtin_prefetch(ap + 32, 0, 3);
    float4 p0 = *(const float4*)(ap);
    float4 p1 = *(const float4*)(ap + 4);
    float4 p2 = *(const float4*)(ap + 8);
    float4 p3 = *(const float4*)(ap + 12);
    v16bf a;
    a[0]  = f2bf(p0.x); a[1]  = f2bf(p0.y); a[2]  = f2bf(p0.z); a[3]  = f2bf(p0.w);
    a[4]  = f2bf(p1.x); a[5]  = f2bf(p1.y); a[6]  = f2bf(p1.z); a[7]  = f2bf(p1.w);
    a[8]  = f2bf(p2.x); a[9]  = f2bf(p2.y); a[10] = f2bf(p2.z); a[11] = f2bf(p2.w);
    a[12] = f2bf(p3.x); a[13] = f2bf(p3.y); a[14] = f2bf(p3.z); a[15] = f2bf(p3.w);

#pragma unroll
    for (int j = 0; j < 4; ++j) {
      if (n0 + j * 16 < N) {
        v16bf b;
        if (TB) {
          const float* bp = bp0 + j * (16 * LDB);
          float4 q0 = *(const float4*)(bp);
          float4 q1 = *(const float4*)(bp + 4);
          float4 q2 = *(const float4*)(bp + 8);
          float4 q3 = *(const float4*)(bp + 12);
          b[0]  = f2bf(q0.x); b[1]  = f2bf(q0.y); b[2]  = f2bf(q0.z); b[3]  = f2bf(q0.w);
          b[4]  = f2bf(q1.x); b[5]  = f2bf(q1.y); b[6]  = f2bf(q1.z); b[7]  = f2bf(q1.w);
          b[8]  = f2bf(q2.x); b[9]  = f2bf(q2.y); b[10] = f2bf(q2.z); b[11] = f2bf(q2.w);
          b[12] = f2bf(q3.x); b[13] = f2bf(q3.y); b[14] = f2bf(q3.z); b[15] = f2bf(q3.w);
        } else {
          const float* bp = bp0 + j * 16;
#pragma unroll
          for (int i = 0; i < 8; ++i) {
            b[2 * i]     = f2bf(bp[(2 * i) * LDB]);
            b[2 * i + 1] = f2bf(bp[(2 * i + 1) * LDB]);
          }
        }
        acc[j] = __builtin_amdgcn_wmma_f32_16x16x32_bf16(false, a, false, b, (short)0, acc[j], false, false);
      }
    }
    ap += 32;
    bp0 += TB ? 32 : 32 * (long)LDB;
  }

#pragma unroll
  for (int j = 0; j < 4; ++j) {
    int nb = n0 + j * 16;
    if (nb < N) {
      int n = nb + lm;
#pragma unroll
      for (int g = 0; g < 8; ++g) {
        int m = m0 + g + (lhi ? 8 : 0);
        if (m < M) {
          float v = acc[j][g] * e.scale;
          if (e.bias) v += e.bias[n];
          if (e.act == 1)      v = gelu_exact(v);
          else if (e.act == 2) v = fmaxf(v, 0.f);
          if (e.mask) {
            long idx = (long)(m + e.rowoff) * e.mld + n;
            if (!e.mask[idx]) v = -1e9f;
            v += e.wvb[idx];
          }
          C[(long)m * ldc + n] = v;
        }
      }
    }
  }
}

// ---------------------------------------------------------------------------
// Projection front-end: time2vec concat + LayerNorm(g,b), zero-padded rows
// ---------------------------------------------------------------------------
__global__ void build_ln_pad(const float* __restrict__ X, int ldx,
                             const float* __restrict__ tsv,
                             const float* __restrict__ tw0, const float* __restrict__ tw,
                             const float* __restrict__ tb,
                             const float* __restrict__ g, const float* __restrict__ bv,
                             float* __restrict__ OUT, int kpad, int M, int base, int D)
{
  int row = blockIdx.x;
  __shared__ float sx[800];
  __shared__ float red[256];
  int tid = threadIdx.x;
  float t = tsv ? tsv[row] : 0.f;
  float lsum = 0.f;
  for (int i = tid; i < D; i += 256) {
    float xi;
    if (i < base)       xi = X[(long)row * ldx + i];
    else if (i == base) xi = tw0[0] * t + tb[0];
    else                xi = sinf(tw[i - base - 1] * t + tb[i - base]);
    sx[i] = xi; lsum += xi;
  }
  float mu; RED_SUM(lsum, red, mu); mu /= (float)D;
  float lv = 0.f;
  for (int i = tid; i < D; i += 256) { float d = sx[i] - mu; lv += d * d; }
  float var; RED_SUM(lv, red, var);
  float rs = rsqrtf(var / (float)D + 1e-5f);
  for (int i = tid; i < kpad; i += 256) {
    float y = 0.f;
    if (i < D) y = (sx[i] - mu) * rs * g[i] + bv[i];
    OUT[(long)row * kpad + i] = y;
  }
}

// ---------------------------------------------------------------------------
// Edge attention (3 passes, segment softmax over destination nodes)
// ---------------------------------------------------------------------------
__global__ void edge_logits(const float* __restrict__ QH, const float* __restrict__ KH,
                            const int* __restrict__ src, const int* __restrict__ dst,
                            const float* __restrict__ R,
                            float* __restrict__ elog, unsigned* __restrict__ segmax,
                            int E, int qoff, int koff)
{
  long gid = (long)blockIdx.x * 256 + threadIdx.x;
  if (gid >= (long)E * NHEAD) return;
  int e = (int)(gid >> 2), h = (int)(gid & 3);
  const float* q = QH + (long)(qoff + dst[e]) * HIDD + h * DKH;
  const float* k = KH + (long)(koff + src[e]) * HIDD + h * DKH;
  const float* r = R + h * DKH;
  float acc = 0.f;
#pragma unroll
  for (int d = 0; d < DKH; ++d) acc += q[d] * (k[d] + r[d]);
  acc *= 0.14433756729740643f;                 // 1/sqrt(48)
  acc = acc > 0.f ? acc : 0.01f * acc;          // leaky_relu
  elog[gid] = acc;
  atomicMax(&segmax[dst[e] * NHEAD + h], ford(acc));
}

__global__ void edge_exp(float* __restrict__ elog, const unsigned* __restrict__ segmax,
                         const int* __restrict__ dst, float* __restrict__ segsum, int E)
{
  long gid = (long)blockIdx.x * 256 + threadIdx.x;
  if (gid >= (long)E * NHEAD) return;
  int e = (int)(gid >> 2), h = (int)(gid & 3);
  float m = funord(segmax[dst[e] * NHEAD + h]);
  float v = expf(elog[gid] - m);
  elog[gid] = v;
  atomicAdd(&segsum[dst[e] * NHEAD + h], v);
}

__global__ void edge_accum(const float* __restrict__ elog, const float* __restrict__ segsum,
                           const float* __restrict__ VH,
                           const int* __restrict__ src, const int* __restrict__ dst,
                           float* __restrict__ MSG, int E, int voff, int moff)
{
  long gid = (long)blockIdx.x * 256 + threadIdx.x;
  if (gid >= (long)E * HIDD) return;
  int e = (int)(gid / HIDD);
  int d = (int)(gid - (long)e * HIDD);
  int h = d / DKH;
  float a = elog[(long)e * NHEAD + h] / (segsum[dst[e] * NHEAD + h] + 1e-16f);
  atomicAdd(&MSG[(long)(moff + dst[e]) * HIDD + d],
            a * VH[(long)(voff + src[e]) * HIDD + d]);
}

__global__ void add_rows(float* __restrict__ A, const float* __restrict__ B, long n) {
  long i = (long)blockIdx.x * 256 + threadIdx.x;
  if (i < n) A[i] += B[i];
}

// ---------------------------------------------------------------------------
// RGT node update: Y = ln(ln(hr)) * g + b   (double layernorm), D=192
// ---------------------------------------------------------------------------
__global__ void rgt_norm(const float* __restrict__ Hin, float* __restrict__ Y,
                         const float* __restrict__ g, const float* __restrict__ b, int M)
{
  int row = blockIdx.x, tid = threadIdx.x;
  __shared__ float red[256];
  float x = (tid < HIDD) ? Hin[(long)row * HIDD + tid] : 0.f;
  float mu; RED_SUM((tid < HIDD) ? x : 0.f, red, mu); mu /= (float)HIDD;
  float d = (tid < HIDD) ? (x - mu) : 0.f;
  float var; RED_SUM(d * d, red, var);
  float z = d * rsqrtf(var / (float)HIDD + 1e-5f);
  float mu2; RED_SUM((tid < HIDD) ? z : 0.f, red, mu2); mu2 /= (float)HIDD;
  float d2 = (tid < HIDD) ? (z - mu2) : 0.f;
  float var2; RED_SUM(d2 * d2, red, var2);
  float rs2 = rsqrtf(var2 / (float)HIDD + 1e-5f);
  if (tid < HIDD) Y[(long)row * HIDD + tid] = d2 * rs2 * g[tid] + b[tid];
}

__global__ void final_norm_relu(float* __restrict__ Hp,
                                const float* __restrict__ g, const float* __restrict__ b, int M)
{
  int row = blockIdx.x, tid = threadIdx.x;
  __shared__ float red[256];
  float x = (tid < HIDD) ? Hp[(long)row * HIDD + tid] : 0.f;
  float mu; RED_SUM((tid < HIDD) ? x : 0.f, red, mu); mu /= (float)HIDD;
  float d = (tid < HIDD) ? (x - mu) : 0.f;
  float var; RED_SUM(d * d, red, var);
  float rs = rsqrtf(var / (float)HIDD + 1e-5f);
  if (tid < HIDD) Hp[(long)row * HIDD + tid] = fmaxf(d * rs * g[tid] + b[tid], 0.f);
}

// ---------------------------------------------------------------------------
// Fused vote-edge MLP: ef = (relu(raw@W1+b1)@W2+b2)*(pol+.01), scatter h_lt*ef
// 16 edges per block; 2 WMMA stages with stage-1 staged in LDS.
// Single-base-pointer B loads with immediate offsets (HIDD is constant).
// ---------------------------------------------------------------------------
__global__ void vote_fused(const float* __restrict__ ea,
                           const float* __restrict__ W1, const float* __restrict__ b1,
                           const float* __restrict__ W2, const float* __restrict__ b2,
                           const float* __restrict__ HLT,
                           const int* __restrict__ vs, const int* __restrict__ vd,
                           float* __restrict__ VMSG, int E)
{
  __shared__ float s1[16 * HIDD];
  int e0 = blockIdx.x * 16;
  int wave = threadIdx.x >> 5, lane = threadIdx.x & 31;
  int lm = lane & 15, lhi = lane >> 4;
  int arow = e0 + lm;
  int arowc = arow < E ? arow : (E - 1);
  const float* earow = ea + (long)arowc * 385 + 1 + (lhi ? 16 : 0);

  for (int nt = wave; nt < 12; nt += 8) {
    v8f acc;
#pragma unroll
    for (int g = 0; g < 8; ++g) acc[g] = 0.f;
    int n0 = nt * 16;
    const float* ar = earow;
    const float* bp = W1 + (long)(lhi ? 16 : 0) * HIDD + n0 + lm;
    for (int k0 = 0; k0 < 384; k0 += 32) {
      v16bf a, b;
#pragma unroll
      for (int i = 0; i < 8; ++i) {
        int k = 2 * i;
        a[2 * i]     = f2bf(ar[k]);
        a[2 * i + 1] = f2bf(ar[k + 1]);
      }
#pragma unroll
      for (int i = 0; i < 8; ++i) {
        b[2 * i]     = f2bf(bp[(2 * i) * HIDD]);
        b[2 * i + 1] = f2bf(bp[(2 * i + 1) * HIDD]);
      }
      acc = __builtin_amdgcn_wmma_f32_16x16x32_bf16(false, a, false, b, (short)0, acc, false, false);
      ar += 32;
      bp += 32 * HIDD;
    }
    int n = n0 + lm;
#pragma unroll
    for (int g = 0; g < 8; ++g) {
      int m = g + (lhi ? 8 : 0);
      s1[m * HIDD + n] = fmaxf(acc[g] + b1[n], 0.f);
    }
  }
  __syncthreads();
  for (int nt = wave; nt < 12; nt += 8) {
    v8f acc;
#pragma unroll
    for (int g = 0; g < 8; ++g) acc[g] = 0.f;
    int n0 = nt * 16;
    const float* sr = s1 + lm * HIDD + (lhi ? 16 : 0);
    const float* bp = W2 + (long)(lhi ? 16 : 0) * HIDD + n0 + lm;
    for (int k0 = 0; k0 < HIDD; k0 += 32) {
      v16bf a, b;
#pragma unroll
      for (int i = 0; i < 8; ++i) {
        int k = 2 * i;
        a[2 * i]     = f2bf(sr[k]);
        a[2 * i + 1] = f2bf(sr[k + 1]);
      }
#pragma unroll
      for (int i = 0; i < 8; ++i) {
        b[2 * i]     = f2bf(bp[(2 * i) * HIDD]);
        b[2 * i + 1] = f2bf(bp[(2 * i + 1) * HIDD]);
      }
      acc = __builtin_amdgcn_wmma_f32_16x16x32_bf16(false, a, false, b, (short)0, acc, false, false);
      sr += 32;
      bp += 32 * HIDD;
    }
    int n = n0 + lm;
#pragma unroll
    for (int g = 0; g < 8; ++g) {
      int m = g + (lhi ? 8 : 0);
      int e = e0 + m;
      if (e < E) {
        float pol = fminf(fmaxf(ea[(long)e * 385], 0.f), 1.f);
        float ef = (acc[g] + b2[n]) * (pol + 0.01f);
        atomicAdd(&VMSG[(long)vd[e] * HIDD + n], ef * HLT[(long)vs[e] * HIDD + n]);
      }
    }
  }
}

// ---------------------------------------------------------------------------
// seg-mean scatter / mix, heads, small helpers
// ---------------------------------------------------------------------------
__global__ void seg_scatter(const float* __restrict__ SRC, const int* __restrict__ map,
                            float* __restrict__ ACC, float* __restrict__ CNT, int M)
{
  long gid = (long)blockIdx.x * 256 + threadIdx.x;
  if (gid >= (long)M * HIDD) return;
  int j = (int)(gid / HIDD);
  int d = (int)(gid - (long)j * HIDD);
  int dn = map[j];
  atomicAdd(&ACC[(long)dn * HIDD + d], SRC[gid]);
  if (d == 0) atomicAdd(&CNT[dn], 1.f);
}
__global__ void seg_mix(float* __restrict__ DST, const float* __restrict__ ACC,
                        const float* __restrict__ CNT, int M)
{
  long gid = (long)blockIdx.x * 256 + threadIdx.x;
  if (gid >= (long)M * HIDD) return;
  int b = (int)(gid / HIDD);
  DST[gid] = 0.7f * DST[gid] + 0.3f * ACC[gid] / fmaxf(CNT[b], 1.f);
}

__global__ void succ_kernel(const float* __restrict__ HB, const float* __restrict__ Wv,
                            const float* __restrict__ bb, float* __restrict__ outs,
                            float* __restrict__ sigs, int M)
{
  int row = blockIdx.x, tid = threadIdx.x;
  __shared__ float red[256];
  float v = (tid < HIDD) ? HB[(long)row * HIDD + tid] * Wv[tid] : 0.f;
  float s; RED_SUM(v, red, s);
  if (tid == 0) {
    float t = s + bb[0];
    outs[row] = t;
    sigs[row] = 1.f / (1.f + expf(-t));
  }
}

__global__ void topic_id_kernel(const float* __restrict__ oh, int* __restrict__ tidv, int M) {
  int b = blockIdx.x * 256 + threadIdx.x;
  if (b >= M) return;
  int id = 0;
  for (int k = 0; k < NKT; ++k) if (oh[(long)b * NKT + k] > 0.5f) id = k;
  tidv[b] = id;
}
__global__ void av2b_kernel(const float* __restrict__ ea, float* __restrict__ out, int n) {
  int i = blockIdx.x * 256 + threadIdx.x;
  if (i < n) out[i] = fminf(fmaxf(ea[i], 0.01f), 100.f);
}
__global__ void l2norm_rows(float* __restrict__ X, int M) {
  int row = blockIdx.x, tid = threadIdx.x;
  __shared__ float red[256];
  float v = (tid < HIDD) ? X[(long)row * HIDD + tid] : 0.f;
  float ss; RED_SUM(v * v, red, ss);
  float inv = 1.f / fmaxf(sqrtf(ss), 1e-12f);
  if (tid < HIDD) X[(long)row * HIDD + tid] = v * inv;
}

__global__ void softmax_rows(float* __restrict__ S, int Wd) {
  long base = (long)blockIdx.x * Wd;
  int tid = threadIdx.x;
  __shared__ float red[256];
  float m = -3.4e38f;
  for (int i = tid; i < Wd; i += 256) m = fmaxf(m, S[base + i]);
  float mm; RED_MAX(m, red, mm);
  float sum = 0.f;
  for (int i = tid; i < Wd; i += 256) { float v = expf(S[base + i] - mm); S[base + i] = v; sum += v; }
  float tot; RED_SUM(sum, red, tot);
  float inv = 1.f / tot;
  for (int i = tid; i < Wd; i += 256) S[base + i] *= inv;
}

__global__ void pab_scatter(const float* __restrict__ SM, const int* __restrict__ ivd,
                            const float* __restrict__ av2b, float* __restrict__ PAB, int rows)
{
  int j = blockIdx.x * 256 + threadIdx.x;
  int r = blockIdx.y;
  if (j >= 6000) return;
  atomicAdd(&PAB[(long)r * 2000 + ivd[j]], SM[(long)r * 6000 + j] * av2b[j]);
}

__global__ void align_out(const float* __restrict__ PAB, const int* __restrict__ tpid,
                          float* __restrict__ out)
{
  int row = blockIdx.x, tid = threadIdx.x;
  __shared__ float bins[NKT];
  __shared__ float red[256];
  if (tid < NKT) bins[tid] = 0.f;
  __syncthreads();
  float s = 0.f;
  for (int b = tid; b < 2000; b += 256) {
    float p = PAB[(long)row * 2000 + b];
    s += p;
    atomicAdd(&bins[tpid[b]], p);
  }
  float tot; RED_SUM(s, red, tot);
  float inv = 1.f / fmaxf(tot, 1e-9f);
  if (tid < NKT) out[(long)row * NKT + tid] = bins[tid] * inv;
}

__global__ void infl_out(const float* __restrict__ PAB, const float* __restrict__ sigs,
                         float* __restrict__ out)
{
  int row = blockIdx.x, tid = threadIdx.x;
  __shared__ float red[256];
  float s = 0.f;
  for (int b = tid; b < 2000; b += 256) s += PAB[(long)row * 2000 + b] * sigs[b];
  float tot; RED_SUM(s, red, tot);
  if (tid == 0) out[row] = tot;
}

// ---------------------------------------------------------------------------
// Host side
// ---------------------------------------------------------------------------
static inline int gblk(long n) { return (int)((n + 255) / 256); }

static void gemm(hipStream_t s, const float* A, int lda, const float* B, int ldb, int tB,
                 float* C, int ldc, int M, int N, int K, int act, int accum,
                 const float* bias, float scale,
                 const unsigned char* mask = nullptr, const float* wvb = nullptr,
                 int mld = 0, int rowoff = 0)
{
  GemmEpi e{act, accum, bias, scale, mask, wvb, mld, rowoff};
  long tiles = (long)((M + 15) / 16) * ((N + 63) / 64);
  int blocks = (int)((tiles + 7) / 8);
  if (tB)             gemm_wmma_t<1, 192><<<blocks, 256, 0, s>>>(A, lda, B, C, ldc, M, N, K, e);
  else if (ldb == 192) gemm_wmma_t<0, 192><<<blocks, 256, 0, s>>>(A, lda, B, C, ldc, M, N, K, e);
  else if (ldb == 768) gemm_wmma_t<0, 768><<<blocks, 256, 0, s>>>(A, lda, B, C, ldc, M, N, K, e);
  else                 gemm_wmma_t<0, 32><<<blocks, 256, 0, s>>>(A, lda, B, C, ldc, M, N, K, e);
}

extern "C" void kernel_launch(void* const* d_in, const int* in_sizes, int n_in,
                              void* d_out, int out_size, void* d_ws, size_t ws_size,
                              hipStream_t stream)
{
  (void)in_sizes; (void)n_in; (void)out_size; (void)ws_size;
  float* wf = (float*)d_ws;
  float* out = (float*)d_out;
  auto F  = [&](int i) { return (const float*)d_in[i]; };
  auto Ii = [&](int i) { return (const int*)d_in[i]; };

  float*    H    = wf + WS_H;
  float*    MSG  = wf + WS_MSG;
  float*    QH   = wf + WS_QH;
  float*    KH   = wf + WS_KH;
  float*    VH   = wf + WS_VH;
  float*    YB   = wf + WS_YB;
  float*    TB   = wf + WS_TB;
  float*    XPAD = wf + WS_XPAD;
  float*    ELOG = wf + WS_ELOG;
  unsigned* SEGM = (unsigned*)(wf + WS_SEGM);
  float*    SEGS = wf + WS_SEGS;
  float*    VMSG = wf + WS_VMSG;
  float*    ACC  = wf + WS_ACC;
  float*    CNT  = wf + WS_CNT;
  float*    AQ   = wf + WS_AQ;
  float*    BK   = wf + WS_BK;
  float*    SIM  = wf + WS_SIM;
  float*    PAB  = wf + WS_PAB;
  float*    SIGS = wf + WS_SIGS;
  float*    AV2B = wf + WS_AV2B;
  int*      TPID = (int*)(wf + WS_TID);
  float*    WPAD = wf + WS_WPAD;

  // ---- 1. per-type projection: h = gelu(ln(concat(x, t2v)) @ W) ----
  struct TypeInfo { int fidx, tsidx, pidx, M, base, D, kpad; long hoff; };
  static const TypeInfo types[7] = {
    {0,  7, 38,  2000, 389, 401, 416, OFF_BILL},
    {1,  8, 41,  6000, 769, 781, 800, OFF_BV},
    {2, -1, 44,  2000, 385, 385, 416, OFF_LEG},
    {3,  9, 47,  8000,   2,  14,  32, OFF_LT},
    {4, -1, 50,   500, 385, 385, 416, OFF_COM},
    {5, -1, 53, 10000, 384, 384, 384, OFF_DON},
    {6, -1, 56,  3000, 384, 384, 384, OFF_LOB},
  };
  for (int t = 0; t < 7; ++t) {
    const TypeInfo& ti = types[t];
    build_ln_pad<<<ti.M, 256, 0, stream>>>(
        F(ti.fidx), ti.base,
        ti.tsidx >= 0 ? F(ti.tsidx) : nullptr,
        F(35), F(36), F(37),
        F(ti.pidx), F(ti.pidx + 1),
        XPAD, ti.kpad, ti.M, ti.base, ti.D);
    pad_weight<<<gblk((long)ti.kpad * HIDD), 256, 0, stream>>>(F(ti.pidx + 2), WPAD, ti.D, ti.kpad);
    gemm(stream, XPAD, ti.kpad, WPAD, HIDD, 0,
         H + ti.hoff * HIDD, HIDD, ti.M, HIDD, ti.kpad, /*gelu*/1, 0, nullptr, 1.f);
  }

  // ---- 2. RGT layers ----
  struct RelInfo { int es, ed, E, soff, doff, ndst; };
  static const RelInfo rels[6] = {
    {21, 22,   6000, OFF_BV,  OFF_BILL, 2000},  // is_version
    {23, 24,   8000, OFF_LEG, OFF_LT,   8000},  // samePerson
    {25, 26, 300000, OFF_LT,  OFF_BV,   6000},  // voted_on
    {27, 28,  40000, OFF_COM, OFF_BILL, 2000},  // reviews
    {29, 30, 100000, OFF_DON, OFF_LEG,  2000},  // donated_to
    {31, 32,  20000, OFF_LOB, OFF_COM,   500},  // lobbied
  };
  for (int L = 0; L < 3; ++L) {
    int rb = 59 + L * 13;
    gemm(stream, H, HIDD, F(rb + 0), HIDD, 0, QH, HIDD, NTOT, HIDD, HIDD, 0, 0, nullptr, 1.f);
    gemm(stream, H, HIDD, F(rb + 1), HIDD, 0, KH, HIDD, NTOT, HIDD, HIDD, 0, 0, nullptr, 1.f);
    gemm(stream, H, HIDD, F(rb + 2), HIDD, 0, VH, HIDD, NTOT, HIDD, HIDD, 0, 0, nullptr, 1.f);
    fill_f32<<<gblk((long)NTOT * HIDD), 256, 0, stream>>>(MSG, 0.f, (long)NTOT * HIDD);
    for (int r = 0; r < 6; ++r) {
      const RelInfo& ri = rels[r];
      fill_u32<<<gblk((long)ri.ndst * NHEAD), 256, 0, stream>>>(SEGM, 0u, (long)ri.ndst * NHEAD);
      fill_f32<<<gblk((long)ri.ndst * NHEAD), 256, 0, stream>>>(SEGS, 0.f, (long)ri.ndst * NHEAD);
      edge_logits<<<gblk((long)ri.E * NHEAD), 256, 0, stream>>>(
          QH, KH, Ii(ri.es), Ii(ri.ed), F(rb + 3 + r), ELOG, SEGM, ri.E, ri.doff, ri.soff);
      edge_exp<<<gblk((long)ri.E * NHEAD), 256, 0, stream>>>(ELOG, SEGM, Ii(ri.ed), SEGS, ri.E);
      edge_accum<<<gblk((long)ri.E * HIDD), 256, 0, stream>>>(
          ELOG, SEGS, VH, Ii(ri.es), Ii(ri.ed), MSG, ri.E, ri.soff, ri.doff);
    }
    add_rows<<<gblk((long)NTOT * HIDD), 256, 0, stream>>>(H, MSG, (long)NTOT * HIDD);
    rgt_norm<<<NTOT, 256, 0, stream>>>(H, YB, F(rb + 9), F(rb + 10), NTOT);
    for (int c0 = 0; c0 < NTOT; c0 += 8192) {
      int rows = NTOT - c0 < 8192 ? NTOT - c0 : 8192;
      gemm(stream, YB + (long)c0 * HIDD, HIDD, F(rb + 11), 768, 0,
           TB, 768, rows, 768, HIDD, /*gelu*/1, 0, nullptr, 1.f);
      gemm(stream, TB, 768, F(rb + 12), HIDD, 0,
           H + (long)c0 * HIDD, HIDD, rows, HIDD, 768, 0, /*accum*/1, nullptr, 1.f);
    }
  }

  // ---- 3. vote edge MLP (fused 2-stage WMMA) ----
  fill_f32<<<gblk(6000L * HIDD), 256, 0, stream>>>(VMSG, 0.f, 6000L * HIDD);
  vote_fused<<<(300000 + 15) / 16, 256, 0, stream>>>(
      F(10), F(98), F(99), F(100), F(101),
      H + (long)OFF_LT * HIDD, Ii(25), Ii(26), VMSG, 300000);
  add_rows<<<gblk(6000L * HIDD), 256, 0, stream>>>(H + (long)OFF_BV * HIDD, VMSG, 6000L * HIDD);

  // ---- 4. final per-type norm + relu ----
  for (int t = 0; t < 7; ++t) {
    const TypeInfo& ti = types[t];
    final_norm_relu<<<ti.M, 256, 0, stream>>>(H + ti.hoff * HIDD, F(102 + 2 * t), F(103 + 2 * t), ti.M);
  }

  // ---- 5. bill / legislator mixing with segment means ----
  fill_f32<<<gblk(2000L * HIDD), 256, 0, stream>>>(ACC, 0.f, 2000L * HIDD);
  fill_f32<<<gblk(2000L), 256, 0, stream>>>(CNT, 0.f, 2000L);
  seg_scatter<<<gblk(6000L * HIDD), 256, 0, stream>>>(H + (long)OFF_BV * HIDD, Ii(22), ACC, CNT, 6000);
  seg_mix<<<gblk(2000L * HIDD), 256, 0, stream>>>(H + (long)OFF_BILL * HIDD, ACC, CNT, 2000);
  fill_f32<<<gblk(2000L * HIDD), 256, 0, stream>>>(ACC, 0.f, 2000L * HIDD);
  fill_f32<<<gblk(2000L), 256, 0, stream>>>(CNT, 0.f, 2000L);
  seg_scatter<<<gblk(8000L * HIDD), 256, 0, stream>>>(H + (long)OFF_LT * HIDD, Ii(23), ACC, CNT, 8000);
  seg_mix<<<gblk(2000L * HIDD), 256, 0, stream>>>(H + (long)OFF_LEG * HIDD, ACC, CNT, 2000);

  // ---- 6. heads: topic logits, success, sigmoids, helpers ----
  gemm(stream, H + (long)OFF_BILL * HIDD, HIDD, F(116), NKT, 0, out, NKT, 2000, NKT, HIDD, 0, 0, nullptr, 1.f);
  succ_kernel<<<2000, 256, 0, stream>>>(H + (long)OFF_BILL * HIDD, F(117), F(118), out + 64000, SIGS, 2000);
  topic_id_kernel<<<gblk(2000L), 256, 0, stream>>>(F(20), TPID, 2000);
  av2b_kernel<<<gblk(6000L), 256, 0, stream>>>(F(11), AV2B, 6000);

  // ---- 7. actor alignment / influence attention ----
  struct ActInfo { int Na; long hoff; int midx, widx; long aoff, ioff; };
  static const ActInfo acts[4] = {
    { 2000, OFF_LEG, 12, 16,  66000, 562000},
    {  500, OFF_COM, 13, 17, 130000, 564000},
    {10000, OFF_DON, 14, 18, 146000, 564500},
    { 3000, OFF_LOB, 15, 19, 466000, 574500},
  };
  const float inv_sqrt_hid = 0.072168783648703216f;   // 1/sqrt(192)
  for (int a = 0; a < 4; ++a) {
    const ActInfo& ai = acts[a];
    for (int ps = 0; ps < 2; ++ps) {
      int pq = (ps == 0 ? 119 : 127) + a * 2;
      gemm(stream, H + ai.hoff * HIDD, HIDD, F(pq), HIDD, 0, AQ, HIDD, ai.Na, HIDD, HIDD, 0, 0, nullptr, 1.f);
      l2norm_rows<<<ai.Na, 256, 0, stream>>>(AQ, ai.Na);
      gemm(stream, H + (long)OFF_BV * HIDD, HIDD, F(pq + 1), HIDD, 0, BK, HIDD, 6000, HIDD, HIDD, 0, 0, nullptr, 1.f);
      l2norm_rows<<<6000, 256, 0, stream>>>(BK, 6000);
      for (int r0 = 0; r0 < ai.Na; r0 += 1024) {
        int rows = ai.Na - r0 < 1024 ? ai.Na - r0 : 1024;
        gemm(stream, AQ + (long)r0 * HIDD, HIDD, BK, HIDD, /*transB*/1,
             SIM, 6000, rows, 6000, HIDD, 0, 0, nullptr, inv_sqrt_hid,
             (const unsigned char*)d_in[ai.midx], F(ai.widx), 6000, r0);
        softmax_rows<<<rows, 256, 0, stream>>>(SIM, 6000);
        fill_f32<<<gblk((long)rows * 2000), 256, 0, stream>>>(PAB, 0.f, (long)rows * 2000);
        pab_scatter<<<dim3((6000 + 255) / 256, rows), 256, 0, stream>>>(SIM, Ii(22), AV2B, PAB, rows);
        if (ps == 0)
          align_out<<<rows, 256, 0, stream>>>(PAB, TPID, out + ai.aoff + (long)r0 * NKT);
        else
          infl_out<<<rows, 256, 0, stream>>>(PAB, SIGS, out + ai.ioff + r0);
      }
    }
  }
}
